// MultiModalReasoningEngine_32263794327948
// MI455X (gfx1250) — matrix operand order repaired and checked by reading, not compiled, not run on hardware
//
#include <hip/hip_runtime.h>
#include <math.h>
#include <stdint.h>

typedef __bf16 bf16_t;
typedef __attribute__((ext_vector_type(16))) __bf16 v16bf;
typedef __attribute__((ext_vector_type(8)))  __bf16 v8bf;
typedef __attribute__((ext_vector_type(4)))  __bf16 v4bf;
typedef __attribute__((ext_vector_type(8)))  float   v8f;

#define Mm 8
#define Bb 4096
#define Dd 512
#define Pp 56

#define BM 128
#define BN 128
#define BK 32
#define LDT 40   // padded LDS row stride (bf16 elems): 80B rows -> conflict-free b128 frag loads

// ---------------------------------------------------------------------------
// Generic batched bf16 GEMM:  C[p] = act( A[p(b)] (rows x K) @ W[p]^T (N x K) + bias[p] )
// Double-buffered LDS tiles filled with CDNA5 async global->LDS DMA
// (global_load_async_to_lds_b128, ASYNCcnt), WMMA bf16 compute.
// Dual output: fp32 (Cf) and/or bf16 (Cb, with its own ld for writing into concat buffers).
// a_tgt_mode=1 maps batch p -> tgt pair index (j<src ? j : j+1).
// ---------------------------------------------------------------------------
__global__ __launch_bounds__(256) void gemm_bf16_kernel(
    const bf16_t* __restrict__ A, long long strideA,
    const bf16_t* __restrict__ W, long long strideW,
    const float*  __restrict__ bias, long long strideBias,
    float* __restrict__ Cf, long long strideCf, int ldCf,
    bf16_t* __restrict__ Cb, long long strideCb, int ldCb,
    int rows, int N, int K, int relu, int a_tgt_mode)
{
    __shared__ __align__(16) bf16_t sA[2][BM * LDT];
    __shared__ __align__(16) bf16_t sB[2][BN * LDT];

    const int p  = blockIdx.z;
    int ab = p;
    if (a_tgt_mode) { int s = p / 7, j = p % 7; ab = (j < s) ? j : j + 1; }
    const bf16_t* Ab = A + (long long)ab * strideA;
    const bf16_t* Wb = W + (long long)p  * strideW;

    const int tid  = threadIdx.x;
    const int lane = tid & 31;
    const int wave = tid >> 5;
    const int wm   = wave & 1;   // 2 waves along M (64 rows each)
    const int wn   = wave >> 1;  // 4 waves along N (32 cols each)
    const int hh   = lane >> 4;  // half-wave id
    const int lr   = lane & 15;

    const int row_block = blockIdx.y * BM;
    const int col_block = blockIdx.x * BN;

    // per-thread staging chunks: ci = tid, tid+256 ; row = ci>>2, col = (ci&3)*8
    const int r0 = tid >> 2,          c0 = (tid & 3) * 8;
    const int r1 = (tid + 256) >> 2,  c1 = ((tid + 256) & 3) * 8;

    // issue one K-tile's async loads into LDS buffer `buf` (4 x b128 per thread)
    auto issue_tile = [&](int k0, int buf) {
        unsigned la0 = (unsigned)(uintptr_t)&sA[buf][r0 * LDT + c0];
        unsigned la1 = (unsigned)(uintptr_t)&sA[buf][r1 * LDT + c1];
        unsigned lb0 = (unsigned)(uintptr_t)&sB[buf][r0 * LDT + c0];
        unsigned lb1 = (unsigned)(uintptr_t)&sB[buf][r1 * LDT + c1];
        const bf16_t* ga0 = &Ab[(long long)(row_block + r0) * K + k0 + c0];
        const bf16_t* ga1 = &Ab[(long long)(row_block + r1) * K + k0 + c1];
        const bf16_t* gb0 = &Wb[(long long)(col_block + r0) * K + k0 + c0];
        const bf16_t* gb1 = &Wb[(long long)(col_block + r1) * K + k0 + c1];
        asm volatile("global_load_async_to_lds_b128 %0, %1, off" :: "v"(la0), "v"(ga0) : "memory");
        asm volatile("global_load_async_to_lds_b128 %0, %1, off" :: "v"(la1), "v"(ga1) : "memory");
        asm volatile("global_load_async_to_lds_b128 %0, %1, off" :: "v"(lb0), "v"(gb0) : "memory");
        asm volatile("global_load_async_to_lds_b128 %0, %1, off" :: "v"(lb1), "v"(gb1) : "memory");
    };

    v8f acc[4][2];
#pragma unroll
    for (int i = 0; i < 4; ++i)
#pragma unroll
        for (int j = 0; j < 2; ++j)
#pragma unroll
            for (int e = 0; e < 8; ++e) acc[i][j][e] = 0.0f;

    const int nt = K / BK;
    issue_tile(0, 0);

    for (int t = 0; t < nt; ++t) {
        const int buf = t & 1;
        if (t + 1 < nt) {
            issue_tile((t + 1) * BK, buf ^ 1);
            // in-order async completion: <=4 outstanding => tile t fully in LDS
            asm volatile("s_wait_asynccnt 0x4" ::: "memory");
        } else {
            asm volatile("s_wait_asynccnt 0x0" ::: "memory");
        }
        __syncthreads();   // all waves' portions of tile t resident

        // A fragments: documented interleaved layout (K 0-7/16-23 | 8-15/24-31)
        v16bf af[4];
#pragma unroll
        for (int i = 0; i < 4; ++i) {
            const bf16_t* base = &sA[buf][(wm * 64 + i * 16 + lr) * LDT + hh * 8];
            v8bf lo = *(const v8bf*)base;
            v8bf hi = *(const v8bf*)(base + 16);
#pragma unroll
            for (int e = 0; e < 8; ++e) { af[i][e] = lo[e]; af[i][8 + e] = hi[e]; }
        }
        // B fragments: interleaved layout (K 0-7/16-23 | 8-15/24-31), N = lane&15
        v16bf bfm[2];
#pragma unroll
        for (int j = 0; j < 2; ++j) {
            const bf16_t* base = &sB[buf][(wn * 32 + j * 16 + lr) * LDT + hh * 8];
            v8bf lo = *(const v8bf*)base;
            v8bf hi = *(const v8bf*)(base + 16);
#pragma unroll
            for (int e = 0; e < 8; ++e) { bfm[j][e] = lo[e]; bfm[j][8 + e] = hi[e]; }
        }
#pragma unroll
        for (int i = 0; i < 4; ++i)
#pragma unroll
            for (int j = 0; j < 2; ++j)
                acc[i][j] = __builtin_amdgcn_wmma_f32_16x16x32_bf16(
                    false, af[i], false, bfm[j], (short)0, acc[i][j], false, false);

        __syncthreads();   // done reading buf before it is refilled at t+2
    }

    // epilogue: C element (VGPR r, lane) -> row = 8*half + r, col = lane&15
#pragma unroll
    for (int i = 0; i < 4; ++i) {
#pragma unroll
        for (int j = 0; j < 2; ++j) {
            int nc = col_block + wn * 32 + j * 16 + lr;
            float bv = bias ? bias[(long long)p * strideBias + nc] : 0.0f;
#pragma unroll
            for (int r = 0; r < 8; ++r) {
                int mr = row_block + wm * 64 + i * 16 + hh * 8 + r;
                float v = acc[i][j][r] + bv;
                if (relu) v = v > 0.0f ? v : 0.0f;
                if (Cf) Cf[(long long)p * strideCf + (long long)mr * ldCf + nc] = v;
                if (Cb) Cb[(long long)p * strideCb + (long long)mr * ldCb + nc] = (bf16_t)v;
            }
        }
    }
}

// ---------------------------------------------------------------------------
// Elementwise / reduction helpers
// ---------------------------------------------------------------------------
__global__ __launch_bounds__(256) void f2bf4_kernel(const float* __restrict__ s,
                                                    bf16_t* __restrict__ d, long long n4)
{
    long long i = (long long)blockIdx.x * blockDim.x + threadIdx.x;
    if (i >= n4) return;
    float4 v = ((const float4*)s)[i];
    v4bf o; o[0] = (bf16_t)v.x; o[1] = (bf16_t)v.y; o[2] = (bf16_t)v.z; o[3] = (bf16_t)v.w;
    ((v4bf*)d)[i] = o;
}

// WvT[p][d][c] = in_proj_w[p][2D + c][d]  (transposed Wv slice -> bf16)
__global__ __launch_bounds__(256) void conv_wvT_kernel(const float* __restrict__ in_proj_w,
                                                       bf16_t* __restrict__ dst)
{
    long long idx = (long long)blockIdx.x * blockDim.x + threadIdx.x;
    if (idx >= (long long)Pp * Dd * Dd) return;
    int p = (int)(idx / (Dd * Dd));
    int r = (int)((idx / Dd) % Dd);   // d
    int c = (int)(idx % Dd);          // c
    dst[idx] = (bf16_t)in_proj_w[(long long)p * 3 * Dd * Dd + (long long)(2 * Dd + c) * Dd + r];
}

__device__ __forceinline__ float wave_sum(float v) {
#pragma unroll
    for (int o = 16; o > 0; o >>= 1) v += __shfl_xor(v, o, 32);
    return v;
}

// bc[p,e] = dot(out_proj_w[p,e,:], in_proj_b[p,2D:3D]) + out_proj_b[p,e]
__global__ __launch_bounds__(256) void bc_kernel(const float* __restrict__ out_w,
                                                 const float* __restrict__ in_b,
                                                 const float* __restrict__ out_b,
                                                 float* __restrict__ bc)
{
    int row = blockIdx.x * 8 + (threadIdx.x >> 5);      // p*D + e
    int lane = threadIdx.x & 31;
    int p = row >> 9, e = row & 511;
    const float* wrow = out_w + ((long long)p * Dd + e) * Dd;
    const float* bv   = in_b + (long long)p * 3 * Dd + 2 * Dd;
    float s = 0.0f;
    for (int c = lane; c < Dd; c += 32) s += wrow[c] * bv[c];
    s = wave_sum(s);
    if (lane == 0) bc[row] = s + out_b[row];
}

// comb[m,b,0:D] = modal_bf; comb[m,b,D:2D] = bf16(mean_j attended[m*7+j,b,:])
__global__ __launch_bounds__(256) void comb_kernel(const bf16_t* __restrict__ modal_bf,
                                                   const float* __restrict__ attended,
                                                   bf16_t* __restrict__ comb)
{
    long long idx = (long long)blockIdx.x * blockDim.x + threadIdx.x; // over M*B*D = 2^24
    if (idx >= (long long)Mm * Bb * Dd) return;
    int m = (int)(idx >> 21);
    long long bd = idx & ((1ll << 21) - 1);   // b*D + d
    long long mb2 = ((long long)m << 21) * 2 + (bd >> 9) * (2 * Dd) + (bd & 511);
    comb[mb2] = modal_bf[idx];
    float s = 0.0f;
#pragma unroll
    for (int j = 0; j < 7; ++j)
        s += attended[(long long)(m * 7 + j) * (Bb * Dd) + bd];
    comb[mb2 + Dd] = (bf16_t)(s * (1.0f / 7.0f));
}

// qf[m,b,0:D] = query_bf[b,:]  (fused half written by GEMM-fus2)
__global__ __launch_bounds__(256) void qf_fill_kernel(const bf16_t* __restrict__ qbf,
                                                      bf16_t* __restrict__ qf)
{
    long long idx = (long long)blockIdx.x * blockDim.x + threadIdx.x;
    if (idx >= (long long)Mm * Bb * Dd) return;
    int m = (int)(idx >> 21);
    long long bd = idx & ((1ll << 21) - 1);
    qf[((long long)m << 22) + (bd >> 9) * (2 * Dd) + (bd & 511)] = qbf[bd];
}

// gate[m,b] = sigmoid(dot(h2[m,b,:], ctrl2_w) + ctrl2_b), one wave per row
__global__ __launch_bounds__(256) void gate_kernel(const bf16_t* __restrict__ h2,
                                                   const float* __restrict__ w2,
                                                   const float* __restrict__ b2,
                                                   float* __restrict__ gate)
{
    int row = blockIdx.x * 8 + (threadIdx.x >> 5);  // m*B + b
    int lane = threadIdx.x & 31;
    float s = 0.0f;
    const bf16_t* hr = h2 + (long long)row * Dd;
    for (int d = lane; d < Dd; d += 32) s += (float)hr[d] * w2[d];
    s = wave_sum(s);
    if (lane == 0) gate[row] = 1.0f / (1.0f + expf(-(s + b2[0])));
}

// unified[b,d] = mean_m fused[m,b,d] * gate[m,b]
__global__ __launch_bounds__(256) void unified_kernel(const float* __restrict__ fused,
                                                      const float* __restrict__ gate,
                                                      float* __restrict__ unified)
{
    long long idx = (long long)blockIdx.x * blockDim.x + threadIdx.x; // over B*D
    if (idx >= (long long)Bb * Dd) return;
    int b = (int)(idx >> 9);
    float s = 0.0f;
#pragma unroll
    for (int m = 0; m < Mm; ++m)
        s += fused[((long long)m << 21) + idx] * gate[m * Bb + b];
    unified[idx] = s * (1.0f / Mm);
}

// importance[b,m] = ||modal_embs[m,b,:]||_2, one wave per (m,b)
__global__ __launch_bounds__(256) void importance_kernel(const float* __restrict__ modal,
                                                         float* __restrict__ out)
{
    int row = blockIdx.x * 8 + (threadIdx.x >> 5);  // m*B + b
    int lane = threadIdx.x & 31;
    int m = row >> 12, b = row & 4095;
    const float* mr = modal + (long long)row * Dd;
    float s = 0.0f;
    for (int d = lane; d < Dd; d += 32) { float x = mr[d]; s += x * x; }
    s = wave_sum(s);
    if (lane == 0) out[(long long)b * Mm + m] = sqrtf(s);
}

__global__ __launch_bounds__(256) void ones_kernel(float* __restrict__ p, long long n)
{
    long long i = (long long)blockIdx.x * blockDim.x + threadIdx.x;
    if (i < n) p[i] = 1.0f;
}

__global__ __launch_bounds__(64) void heatmap_kernel(float* __restrict__ p)
{
    int i = threadIdx.x;
    if (i < Mm * Mm) p[i] = ((i / Mm) == (i % Mm)) ? 0.0f : 1.0f;
}

// ---------------------------------------------------------------------------
extern "C" void kernel_launch(void* const* d_in, const int* in_sizes, int n_in,
                              void* d_out, int out_size, void* d_ws, size_t ws_size,
                              hipStream_t stream)
{
    (void)in_sizes; (void)n_in; (void)out_size; (void)ws_size;
    const float* modal_embs = (const float*)d_in[0];
    const float* query      = (const float*)d_in[1];
    const float* in_proj_w  = (const float*)d_in[2];
    const float* in_proj_b  = (const float*)d_in[3];
    const float* out_proj_w = (const float*)d_in[4];
    const float* out_proj_b = (const float*)d_in[5];
    const float* fus1_w     = (const float*)d_in[6];
    const float* fus1_b     = (const float*)d_in[7];
    const float* fus2_w     = (const float*)d_in[8];
    const float* fus2_b     = (const float*)d_in[9];
    const float* ctrl1_w    = (const float*)d_in[10];
    const float* ctrl1_b    = (const float*)d_in[11];
    const float* ctrl2_w    = (const float*)d_in[12];
    const float* ctrl2_b    = (const float*)d_in[13];

    // output layout (flat fp32 concat, reference return order)
    float* fused_out      = (float*)d_out;                                  // M*B*D
    float* unified_out    = fused_out + (long long)Mm * Bb * Dd;            // B*D
    float* attnw_out      = unified_out + (long long)Bb * Dd;               // P*B
    float* attended_out   = attnw_out + (long long)Pp * Bb;                 // P*B*D
    float* importance_out = attended_out + (long long)Pp * Bb * Dd;         // B*M
    float* heatmap_out    = importance_out + (long long)Bb * Mm;            // M*M

    // workspace carve-out
    char* w = (char*)d_ws;
    size_t off = 0;
    auto alloc = [&](size_t bytes) { void* q = w + off; off += (bytes + 255) & ~(size_t)255; return q; };
    bf16_t* modal_bf = (bf16_t*)alloc((size_t)Mm * Bb * Dd * 2);
    bf16_t* query_bf = (bf16_t*)alloc((size_t)Bb * Dd * 2);
    bf16_t* Wo_bf    = (bf16_t*)alloc((size_t)Pp * Dd * Dd * 2);
    bf16_t* WvT_bf   = (bf16_t*)alloc((size_t)Pp * Dd * Dd * 2);
    bf16_t* Wc_bf    = (bf16_t*)alloc((size_t)Pp * Dd * Dd * 2);
    float*  bc_f     = (float*) alloc((size_t)Pp * Dd * 4);
    bf16_t* fus1_bf  = (bf16_t*)alloc((size_t)Mm * Dd * 2 * Dd * 2);
    bf16_t* fus2_bf  = (bf16_t*)alloc((size_t)Mm * Dd * Dd * 2);
    bf16_t* ctrl1_bf = (bf16_t*)alloc((size_t)Dd * 2 * Dd * 2);
    bf16_t* comb_bf  = (bf16_t*)alloc((size_t)Mm * Bb * 2 * Dd * 2);
    bf16_t* h1_bf    = (bf16_t*)alloc((size_t)Mm * Bb * Dd * 2);
    bf16_t* qf_bf    = (bf16_t*)alloc((size_t)Mm * Bb * 2 * Dd * 2);
    bf16_t* h2_bf    = (bf16_t*)alloc((size_t)Mm * Bb * Dd * 2);
    float*  gate_f   = (float*) alloc((size_t)Mm * Bb * 4);

    auto cvt = [&](const float* s, bf16_t* d, long long n) {
        long long n4 = n / 4;
        f2bf4_kernel<<<(unsigned)((n4 + 255) / 256), 256, 0, stream>>>(s, d, n4);
    };
    // 1) fp32 -> bf16 conversions
    cvt(modal_embs, modal_bf, (long long)Mm * Bb * Dd);
    cvt(query,      query_bf, (long long)Bb * Dd);
    cvt(out_proj_w, Wo_bf,    (long long)Pp * Dd * Dd);
    cvt(fus1_w,     fus1_bf,  (long long)Mm * Dd * 2 * Dd);
    cvt(fus2_w,     fus2_bf,  (long long)Mm * Dd * Dd);
    cvt(ctrl1_w,    ctrl1_bf, (long long)Dd * 2 * Dd);
    conv_wvT_kernel<<<(unsigned)(((long long)Pp * Dd * Dd + 255) / 256), 256, 0, stream>>>(in_proj_w, WvT_bf);

    // 2) folded bias bc = Wo @ bv + bo
    bc_kernel<<<(Pp * Dd) / 8, 256, 0, stream>>>(out_proj_w, in_proj_b, out_proj_b, bc_f);

    // 3) Wc[p] = Wo[p] @ Wv[p]   (rows=512, N=512, K=512, batch=56)
    gemm_bf16_kernel<<<dim3(Dd / BN, Dd / BM, Pp), 256, 0, stream>>>(
        Wo_bf, (long long)Dd * Dd, WvT_bf, (long long)Dd * Dd,
        nullptr, 0, nullptr, 0, 0, Wc_bf, (long long)Dd * Dd, Dd,
        Dd, Dd, Dd, 0, 0);

    // 4) attended[p] = tgt_e[p] @ Wc[p]^T + bc[p]   (attn is identity: softmax over size-1 axis)
    gemm_bf16_kernel<<<dim3(Dd / BN, Bb / BM, Pp), 256, 0, stream>>>(
        modal_bf, (long long)Bb * Dd, Wc_bf, (long long)Dd * Dd,
        bc_f, Dd, attended_out, (long long)Bb * Dd, Dd, nullptr, 0, 0,
        Bb, Dd, Dd, 0, /*tgt map*/1);

    // 5) comb = [modal | mean_pairs(attended)]
    comb_kernel<<<(unsigned)(((long long)Mm * Bb * Dd + 255) / 256), 256, 0, stream>>>(
        modal_bf, attended_out, comb_bf);

    // 6) h1 = relu(comb @ fus1_w^T + fus1_b)   (K=1024)
    gemm_bf16_kernel<<<dim3(Dd / BN, Bb / BM, Mm), 256, 0, stream>>>(
        comb_bf, (long long)Bb * 2 * Dd, fus1_bf, (long long)Dd * 2 * Dd,
        fus1_b, Dd, nullptr, 0, 0, h1_bf, (long long)Bb * Dd, Dd,
        Bb, Dd, 2 * Dd, 1, 0);

    // 7) fused = h1 @ fus2_w^T + fus2_b  -> fp32 output + bf16 into qf[:, D:2D]
    gemm_bf16_kernel<<<dim3(Dd / BN, Bb / BM, Mm), 256, 0, stream>>>(
        h1_bf, (long long)Bb * Dd, fus2_bf, (long long)Dd * Dd,
        fus2_b, Dd, fused_out, (long long)Bb * Dd, Dd,
        qf_bf + Dd, (long long)Bb * 2 * Dd, 2 * Dd,
        Bb, Dd, Dd, 0, 0);

    // 8) qf[:, 0:D] = broadcast(reasoning_query)
    qf_fill_kernel<<<(unsigned)(((long long)Mm * Bb * Dd + 255) / 256), 256, 0, stream>>>(
        query_bf, qf_bf);

    // 9) h2 = relu(qf @ ctrl1_w^T + ctrl1_b)  (shared weight, K=1024)
    gemm_bf16_kernel<<<dim3(Dd / BN, Bb / BM, Mm), 256, 0, stream>>>(
        qf_bf, (long long)Bb * 2 * Dd, ctrl1_bf, 0,
        ctrl1_b, 0, nullptr, 0, 0, h2_bf, (long long)Bb * Dd, Dd,
        Bb, Dd, 2 * Dd, 1, 0);

    // 10) gate + unified
    gate_kernel<<<(Mm * Bb) / 8, 256, 0, stream>>>(h2_bf, ctrl2_w, ctrl2_b, gate_f);
    unified_kernel<<<(unsigned)(((long long)Bb * Dd + 255) / 256), 256, 0, stream>>>(
        fused_out, gate_f, unified_out);

    // 11) trivial outputs: attn_w == 1, importance = row norms, heatmap = 1 off-diag
    ones_kernel<<<(unsigned)(((long long)Pp * Bb + 255) / 256), 256, 0, stream>>>(
        attnw_out, (long long)Pp * Bb);
    importance_kernel<<<(Mm * Bb) / 8, 256, 0, stream>>>(modal_embs, importance_out);
    heatmap_kernel<<<1, 64, 0, stream>>>(heatmap_out);
}
